// FPModule_21809843929148
// MI455X (gfx1250) — compile-verified, matrix operand order checked by reading, not run on hardware
//
#include <hip/hip_runtime.h>
#include <hip/hip_bf16.h>

// ---------------------------------------------------------------------------
// Problem constants (from reference)
// ---------------------------------------------------------------------------
#define N_SRC   4096
#define N_DST   16384
#define C_IN    256
#define C_SKIP  128
#define C_CAT   384      // C_IN + C_SKIP
#define HID     512
#define KNN_K   3
#define BOFF    1000.0f  // BATCH_OFFSET

typedef __attribute__((ext_vector_type(16))) _Float16 v16h;
typedef __attribute__((ext_vector_type(8)))  _Float16 v8h;
typedef __attribute__((ext_vector_type(8)))  float    v8f;

union F16x16 { v16h v; v8h p[2]; };

// ---------------------------------------------------------------------------
// Kernel 1: kNN(k=3) with batch offset; emits normalized inverse-d2 weights.
// Source cloud (4096 pts, 48KB) staged in LDS once per block.
// ---------------------------------------------------------------------------
__global__ __launch_bounds__(256) void knn_kernel(
    const float* __restrict__ pos, const int* __restrict__ batch,
    const float* __restrict__ pos_skip, const int* __restrict__ batch_skip,
    float* __restrict__ wn, int* __restrict__ idx) {
  __shared__ float sx[N_SRC], sy[N_SRC], sz[N_SRC];
  for (int i = threadIdx.x; i < N_SRC; i += 256) {
    float off = (float)batch[i] * BOFF;
    sx[i] = pos[i * 3 + 0] + off;
    sy[i] = pos[i * 3 + 1] + off;
    sz[i] = pos[i * 3 + 2] + off;
  }
  __syncthreads();

  int d = blockIdx.x * 256 + threadIdx.x;   // d < N_DST by grid sizing
  float offq = (float)batch_skip[d] * BOFF;
  float qx = pos_skip[d * 3 + 0] + offq;
  float qy = pos_skip[d * 3 + 1] + offq;
  float qz = pos_skip[d * 3 + 2] + offq;

  float d0 = 3.4e38f, d1 = 3.4e38f, d2 = 3.4e38f;
  int   i0 = 0,       i1 = 0,       i2 = 0;
  for (int s = 0; s < N_SRC; ++s) {
    float dx = qx - sx[s], dy = qy - sy[s], dz = qz - sz[s];
    float dd = dx * dx + dy * dy + dz * dz;
    if (dd < d2) {
      if (dd < d1) {
        if (dd < d0) { d2 = d1; i2 = i1; d1 = d0; i1 = i0; d0 = dd; i0 = s; }
        else         { d2 = d1; i2 = i1; d1 = dd; i1 = s; }
      } else         { d2 = dd; i2 = s; }
    }
  }
  float w0 = 1.0f / fmaxf(d0, 1e-16f);
  float w1 = 1.0f / fmaxf(d1, 1e-16f);
  float w2 = 1.0f / fmaxf(d2, 1e-16f);
  float inv = 1.0f / (w0 + w1 + w2);
  wn[d * 3 + 0] = w0 * inv;  idx[d * 3 + 0] = i0;
  wn[d * 3 + 1] = w1 * inv;  idx[d * 3 + 1] = i1;
  wn[d * 3 + 2] = w2 * inv;  idx[d * 3 + 2] = i2;
}

// ---------------------------------------------------------------------------
// Kernel 2: weight transpose + f32->f16 (W1 [384,512] -> W1t [512,384];
//           W2 [512,384] -> W2t [384,512]).  N-major so B fragments are
//           two contiguous 16B loads per lane.
// ---------------------------------------------------------------------------
__global__ __launch_bounds__(256) void wconv_kernel(
    const float* __restrict__ W1, const float* __restrict__ W2,
    _Float16* __restrict__ W1t, _Float16* __restrict__ W2t) {
  int gid = blockIdx.x * 256 + threadIdx.x;    // total 2*384*512 = 393216
  if (gid < HID * C_CAT) {
    int n = gid / C_CAT, k = gid % C_CAT;
    W1t[gid] = (_Float16)W1[k * HID + n];
  } else {
    int g = gid - HID * C_CAT;
    int n = g / HID, k = g % HID;
    W2t[g] = (_Float16)W2[k * C_CAT + n];
  }
}

// ---------------------------------------------------------------------------
// Kernel 3: build h = concat(knn_interp(x), x_skip): fp32 (residual) + f16 (A op)
// ---------------------------------------------------------------------------
__global__ __launch_bounds__(256) void build_h_kernel(
    const float* __restrict__ x, const float* __restrict__ x_skip,
    const float* __restrict__ wn, const int* __restrict__ idx,
    float* __restrict__ h32, _Float16* __restrict__ h16) {
  int gid = blockIdx.x * 256 + threadIdx.x;    // N_DST*C_CAT threads
  int i = gid / C_CAT, c = gid % C_CAT;
  float v;
  if (c < C_IN) {
    int j0 = idx[i * 3 + 0], j1 = idx[i * 3 + 1], j2 = idx[i * 3 + 2];
    v = wn[i * 3 + 0] * x[j0 * C_IN + c]
      + wn[i * 3 + 1] * x[j1 * C_IN + c]
      + wn[i * 3 + 2] * x[j2 * C_IN + c];
  } else {
    v = x_skip[i * C_SKIP + (c - C_IN)];
  }
  h32[gid] = v;
  h16[gid] = (_Float16)v;
}

// ---------------------------------------------------------------------------
// WMMA fragment loaders (row-major A [M,K]; N-major B [N,K]) — identical
// per-lane layout: lane L holds row (L&15); halves 0..7 <-> K = 8*(L>>4)+0..7,
// halves 8..15 <-> K = 16+8*(L>>4)+0..7  => two b128 loads per fragment.
// ---------------------------------------------------------------------------
__device__ __forceinline__ v16h load_frag(const _Float16* base) {
  F16x16 f;
  f.p[0] = *(const v8h*)(base);
  f.p[1] = *(const v8h*)(base + 16);
  return f.v;
}

// ---------------------------------------------------------------------------
// Kernel 4: GEMM1  a16[16384,512] = relu(h16 @ W1t^T + b1), f16 out.
// Block = 8 waves; block owns 16 rows; wave w owns columns [64w, 64w+64).
// ---------------------------------------------------------------------------
__global__ __launch_bounds__(256) void gemm1_kernel(
    const _Float16* __restrict__ h16, const _Float16* __restrict__ W1t,
    const float* __restrict__ b1, _Float16* __restrict__ a16) {
  const int lane = threadIdx.x & 31;
  const int wave = threadIdx.x >> 5;
  const int m0 = blockIdx.x * 16;
  const int n0 = wave * 64;
  const int r = lane & 15;          // A row / B column / D column (per lane)
  const int g = lane >> 4;          // lane-group: K-phase / D row offset
  const _Float16* Abase = h16 + (size_t)(m0 + r) * C_CAT + 8 * g;

  v8f acc[4] = {};
#pragma unroll
  for (int kk = 0; kk < C_CAT / 32; ++kk) {
    const int k0 = kk * 32;
    v16h a = load_frag(Abase + k0);
#pragma unroll
    for (int t = 0; t < 4; ++t) {
      const _Float16* Bbase = W1t + (size_t)(n0 + t * 16 + r) * C_CAT + k0 + 8 * g;
      v16h b = load_frag(Bbase);
      acc[t] = __builtin_amdgcn_wmma_f32_16x16x32_f16(
          false, a, false, b, (short)0, acc[t], false, false);
    }
  }
#pragma unroll
  for (int t = 0; t < 4; ++t) {
    const int n = n0 + t * 16 + r;
    const float bias = b1[n];
#pragma unroll
    for (int e = 0; e < 8; ++e) {
      const int m = m0 + e + 8 * g;
      float v = acc[t][e] + bias;
      v = v > 0.0f ? v : 0.0f;                 // ReLU
      a16[(size_t)m * HID + n] = (_Float16)v;
    }
  }
}

// ---------------------------------------------------------------------------
// Kernel 5: GEMM2  out[16384,384] = a16 @ W2t^T + b2 + h32 (fused residual).
// Block = 8 waves; wave w owns columns [48w, 48w+48) of this 16-row strip.
// ---------------------------------------------------------------------------
__global__ __launch_bounds__(256) void gemm2_kernel(
    const _Float16* __restrict__ a16, const _Float16* __restrict__ W2t,
    const float* __restrict__ b2, const float* __restrict__ h32,
    float* __restrict__ out) {
  const int lane = threadIdx.x & 31;
  const int wave = threadIdx.x >> 5;
  const int m0 = blockIdx.x * 16;
  const int n0 = wave * 48;
  const int r = lane & 15;
  const int g = lane >> 4;
  const _Float16* Abase = a16 + (size_t)(m0 + r) * HID + 8 * g;

  v8f acc[3] = {};
#pragma unroll
  for (int kk = 0; kk < HID / 32; ++kk) {
    const int k0 = kk * 32;
    v16h a = load_frag(Abase + k0);
#pragma unroll
    for (int t = 0; t < 3; ++t) {
      const _Float16* Bbase = W2t + (size_t)(n0 + t * 16 + r) * HID + k0 + 8 * g;
      v16h b = load_frag(Bbase);
      acc[t] = __builtin_amdgcn_wmma_f32_16x16x32_f16(
          false, a, false, b, (short)0, acc[t], false, false);
    }
  }
#pragma unroll
  for (int t = 0; t < 3; ++t) {
    const int n = n0 + t * 16 + r;
    const float bias = b2[n];
#pragma unroll
    for (int e = 0; e < 8; ++e) {
      const int m = m0 + e + 8 * g;
      out[(size_t)m * C_CAT + n] = acc[t][e] + bias + h32[(size_t)m * C_CAT + n];
    }
  }
}

// ---------------------------------------------------------------------------
// Kernel 6: tuple tail — pos_skip then batch_skip, flattened after `out`.
// ---------------------------------------------------------------------------
__global__ __launch_bounds__(256) void tail_kernel(
    const float* __restrict__ pos_skip, const int* __restrict__ batch_skip,
    float* __restrict__ out_tail) {
  int gid = blockIdx.x * 256 + threadIdx.x;   // N_DST*3 + N_DST threads
  if (gid < N_DST * 3) out_tail[gid] = pos_skip[gid];
  else                 out_tail[gid] = (float)batch_skip[gid - N_DST * 3];
}

// ---------------------------------------------------------------------------
// Host-side orchestration
// ---------------------------------------------------------------------------
extern "C" void kernel_launch(void* const* d_in, const int* in_sizes, int n_in,
                              void* d_out, int out_size, void* d_ws, size_t ws_size,
                              hipStream_t stream) {
  const float* x          = (const float*)d_in[0];
  const float* pos        = (const float*)d_in[1];
  const int*   batch      = (const int*)  d_in[2];
  const float* x_skip     = (const float*)d_in[3];
  const float* pos_skip   = (const float*)d_in[4];
  const int*   batch_skip = (const int*)  d_in[5];
  const float* W1         = (const float*)d_in[6];
  const float* b1         = (const float*)d_in[7];
  const float* W2         = (const float*)d_in[8];
  const float* b2         = (const float*)d_in[9];
  float* out = (float*)d_out;

  char* ws = (char*)d_ws;
  size_t off = 0;
  float*     wn   = (float*)    (ws + off); off += (size_t)N_DST * 3 * 4;
  int*       idx  = (int*)      (ws + off); off += (size_t)N_DST * 3 * 4;
  _Float16*  W1t  = (_Float16*) (ws + off); off += (size_t)HID * C_CAT * 2;
  _Float16*  W2t  = (_Float16*) (ws + off); off += (size_t)C_CAT * HID * 2;
  _Float16*  h16  = (_Float16*) (ws + off); off += (size_t)N_DST * C_CAT * 2;
  _Float16*  a16  = (_Float16*) (ws + off); off += (size_t)N_DST * HID * 2;
  float*     h32  = (float*)    (ws + off); off += (size_t)N_DST * C_CAT * 4;
  (void)ws_size; (void)in_sizes; (void)n_in; (void)out_size;

  // 1) kNN weights/indices
  knn_kernel<<<N_DST / 256, 256, 0, stream>>>(pos, batch, pos_skip, batch_skip, wn, idx);
  // 2) weight transpose + f16 convert
  wconv_kernel<<<(2 * HID * C_CAT) / 256, 256, 0, stream>>>(W1, W2, W1t, W2t);
  // 3) h = concat(interp, x_skip)  (fp32 + f16 copies)
  build_h_kernel<<<(N_DST * C_CAT) / 256, 256, 0, stream>>>(x, x_skip, wn, idx, h32, h16);
  // 4) GEMM1 + ReLU  (WMMA f16 -> f32 acc)
  gemm1_kernel<<<N_DST / 16, 256, 0, stream>>>(h16, W1t, b1, a16);
  // 5) GEMM2 + bias + residual  (WMMA f16 -> f32 acc)
  gemm2_kernel<<<N_DST / 16, 256, 0, stream>>>(a16, W2t, b2, h32, out);
  // 6) tuple tail: pos_skip, batch_skip
  tail_kernel<<<(N_DST * 4) / 256, 256, 0, stream>>>(pos_skip, batch_skip,
                                                     out + (size_t)N_DST * C_CAT);
}
